// GTrXL_42769284334233
// MI455X (gfx1250) — compile-verified, matrix operand order checked by reading, not run on hardware
//
// GTrXL forward for MI455X (gfx1250, wave32, WMMA + Tensor Data Mover).
// All dense math via v_wmma_f32_16x16x32_bf16 (fp32 accumulate); LN/softmax/GRU fp32.
// Compute-bound (~1.2 TFLOP vs ~10us of HBM traffic at 23.3 TB/s), so everything
// funnels through one LDS-double-buffered 128x128x32 WMMA GEMM with fused epilogues.
// B operands are bf16 [N,K], except v in attn@v ([K,N]) (weights pre-transposed once per launch; kv / r_
// written bf16 straight from their projection GEMMs) and [N,K] ones are staged into LDS by the
// TDM (tensor_load_to_lds, TENSORcnt) while the other waves run WMMA on the live
// buffer. Descriptor layout per CDNA5 ISA section 8 (compile-only, not HW-verified).
#include <hip/hip_runtime.h>

// ---------------- problem constants ----------------
constexpr int Lq  = 3;
constexpr int Dm  = 1024;
constexpr int Hh  = 8;
constexpr int HDd = 128;
constexpr int CUR = 512, MEM = 512, BSz = 16;
constexpr int FULL  = CUR + MEM;      // 1024
constexpr int NCUR  = CUR * BSz;      // 8192 rows
constexpr int NFULL = FULL * BSz;     // 16384 rows
constexpr int HDm   = Hh * HDd;       // 1024

typedef __bf16 bf16_t;
typedef __bf16 bf16x16 __attribute__((ext_vector_type(16)));
typedef float  f32x8   __attribute__((ext_vector_type(8)));
typedef unsigned int u32x4 __attribute__((ext_vector_type(4)));
typedef int          i32x4 __attribute__((ext_vector_type(4)));
typedef int          i32x8 __attribute__((ext_vector_type(8)));

#if __has_builtin(__builtin_amdgcn_tensor_load_to_lds)
#define HAVE_TDM 1
#endif

static __device__ __forceinline__ bf16_t f2bf(float x) {
  return (bf16_t)x;   // v_cvt_pk_bf16_f32 (RNE) on gfx1250
}

// ---------------- block reduction helper ----------------
static __device__ __forceinline__ float block_red(float v, float* sm, bool domax) {
  const int t = threadIdx.x;
  sm[t] = v;
  __syncthreads();
  #pragma unroll
  for (int s = 128; s > 0; s >>= 1) {
    if (t < s) sm[t] = domax ? fmaxf(sm[t], sm[t + s]) : (sm[t] + sm[t + s]);
    __syncthreads();
  }
  float r = sm[0];
  __syncthreads();
  return r;
}

// ======================================================================
// Batched WMMA GEMM:  C[z] = act( A[z] @ Bop[z] (+C[z] if ACCUM) + biasScale*bias )
//   A: fp32 row-major [M,K], row stride lda, batch offset z*sA
//   B: bf16. BT=true : row-major [N,K] (ldb = element row stride) -> TDM staged
//            BT=false: row-major [K,N] (values for attn@v) -> transpose staged
//      batch offset (z/bdiv)*sB1 + (z%bdiv)*sB2
//   C: [M,N] row stride ldc, batch offset z*sC; fp32 (CBF=0) or bf16 (CBF=1)
//   ACT: 0=none 1=relu 2=sigmoid
// Block = 256 threads = 8 waves; 128x128 C tile; K step 32.
// Wave grid 4x2, wave tile 32x64 => 2x4 accumulators of v_wmma_f32_16x16x32_bf16.
// ======================================================================
constexpr int BM = 128, BN = 128, BK = 32, LDK = BK + 8;  // LDS row = 80 B

template <bool BT, int ACT, bool ACCUM, bool CBF>
__global__ __launch_bounds__(256)
void gemm_wmma_k(const float* __restrict__ A, const bf16_t* __restrict__ B,
                 void* __restrict__ Cv, const float* __restrict__ bias,
                 float biasScale, int M, int N, int K,
                 int lda, long long sA,
                 int ldb, long long sB1, long long sB2, int bdiv,
                 int ldc, long long sC) {
  __shared__ bf16_t As[2][BM * LDK];
  __shared__ bf16_t Bs[2][BN * LDK];

  const int tid = threadIdx.x;
  const int z   = blockIdx.z;
  A += (long long)z * sA;
  const bf16_t* Bp = B + (long long)(z / bdiv) * sB1 + (long long)(z % bdiv) * sB2;
  float*  Cf = (float*)Cv  + (long long)z * sC;
  bf16_t* Cb = (bf16_t*)Cv + (long long)z * sC;
  const int m0 = blockIdx.y * BM;
  const int n0 = blockIdx.x * BN;

  auto stage = [&](int buf, int k0) {
    { // A: 128x32 fp32 -> bf16 LDS [m][k]; batch all loads, then convert+store
      const int r = tid >> 3, c = (tid & 7) * 4;
      float4 fa[4];
      #pragma unroll
      for (int p = 0; p < 4; ++p)
        fa[p] = *(const float4*)(A + (size_t)(m0 + p * 32 + r) * lda + k0 + c);
      #pragma unroll
      for (int p = 0; p < 4; ++p) {
        union { bf16_t e[4]; uint2 w; } t;
        t.e[0] = f2bf(fa[p].x); t.e[1] = f2bf(fa[p].y);
        t.e[2] = f2bf(fa[p].z); t.e[3] = f2bf(fa[p].w);
        *(uint2*)&As[buf][(p * 32 + r) * LDK + c] = t.w;
      }
    }
    if constexpr (BT) {
#ifdef HAVE_TDM
      // Tensor Data Mover: DMA the 128x32 bf16 tile (rows of 64 B) into Bs[buf],
      // padding 16 B after every 64 B so rows land at our LDS stride of 80 B.
      if (tid < 32) {
        const unsigned lds = (unsigned)(size_t)&Bs[buf][0];
        const unsigned long long ga =
            (unsigned long long)(const void*)Bp + ((size_t)n0 * ldb + k0) * 2ull;
        u32x4 g0; i32x8 g1; i32x4 g2 = {}; i32x4 g3 = {};
        g0[0] = 1u;                                   // count=1 (valid user D#)
        g0[1] = lds;                                  // lds_addr (bytes)
        g0[2] = (unsigned)ga;                         // global_addr[31:0]
        g0[3] = (unsigned)((ga >> 32) & 0x1FFFFFFull) // global_addr[56:32]
              | (2u << 30);                           // type = 2 ("image")
        g1[0] = (1 << 16)     // data_size = 2 bytes
              | (1 << 20)     // pad_enable
              | (3 << 22)     // pad_interval: 16 DWORDs (64 B)
              | (3 << 25);    // pad_amount: 4 DWORDs (16 B)
        g1[1] = (K & 0xFFFF) << 16;                          // tensor_dim0 lo
        g1[2] = ((K >> 16) & 0xFFFF) | ((N & 0xFFFF) << 16); // dim0 hi | tensor_dim1 lo
        g1[3] = ((N >> 16) & 0xFFFF) | (BK << 16);           // dim1 hi | tile_dim0 = 32
        g1[4] = BN;                                          // tile_dim1 = 128 rows
        g1[5] = ldb;                                         // tensor_dim0_stride lo
        g1[6] = 0;
        g1[7] = 0;
#if __clang_major__ >= 23
        i32x8 g4 = {};
        __builtin_amdgcn_tensor_load_to_lds(g0, g1, g2, g3, g4, 0);
#else
        __builtin_amdgcn_tensor_load_to_lds(g0, g1, g2, g3, 0);
#endif
      }
#else
      // fallback: pure 16B copy into Bs[n][k]
      const int n = tid >> 2, c = (tid & 3) * 8;
      #pragma unroll
      for (int p = 0; p < 2; ++p) {
        const int nn = p * 64 + n;
        *(uint4*)&Bs[buf][nn * LDK + c] =
            *(const uint4*)(Bp + (size_t)(n0 + nn) * ldb + k0 + c);
      }
#endif
    } else {
      // bf16 [K,N] -> transpose into Bs[n][k]
      const int k = tid >> 4, nb = (tid & 15) * 8;
      #pragma unroll
      for (int p = 0; p < 2; ++p) {
        const int kk = p * 16 + k;
        union { uint4 r4; bf16_t e[8]; } t;
        t.r4 = *(const uint4*)(Bp + (size_t)(k0 + kk) * ldb + n0 + nb);
        #pragma unroll
        for (int j = 0; j < 8; ++j) Bs[buf][(nb + j) * LDK + kk] = t.e[j];
      }
    }
  };

  auto tdm_wait = [&] {
#ifdef HAVE_TDM
    if constexpr (BT) {
      if (tid < 32) __builtin_amdgcn_s_wait_tensorcnt(0);
    }
#endif
  };

  const int wave = tid >> 5, lane = tid & 31;
  const int wr = wave >> 1, wc = wave & 1;   // 4x2 wave grid
  const int lm = lane & 15;
  const int kh = (lane >> 4) * 8;            // K half select per bf16 A/B layout

  f32x8 acc[2][4] = {};
  union FragU { bf16x16 v; uint4 q[2]; };

  stage(0, 0);
  tdm_wait();
  __syncthreads();
  const int kTiles = K / BK;
  for (int kt = 0; kt < kTiles; ++kt) {
    const int cb = kt & 1;
    if (kt + 1 < kTiles) stage(cb ^ 1, (kt + 1) * BK);
    if (kt + 2 < kTiles)  // speculative L2 prefetch of the A stream, 2 tiles ahead
      __builtin_prefetch(A + (size_t)(m0 + (tid >> 3)) * lda + (kt + 2) * BK, 0, 0);
    FragU a[2], b[4];
    #pragma unroll
    for (int i = 0; i < 2; ++i) {
      const bf16_t* p = &As[cb][(wr * 32 + i * 16 + lm) * LDK + kh];
      a[i].q[0] = *(const uint4*)p;
      a[i].q[1] = *(const uint4*)(p + 16);
    }
    #pragma unroll
    for (int j = 0; j < 4; ++j) {
      const bf16_t* p = &Bs[cb][(wc * 64 + j * 16 + lm) * LDK + kh];
      b[j].q[0] = *(const uint4*)p;
      b[j].q[1] = *(const uint4*)(p + 16);
    }
    #pragma unroll
    for (int i = 0; i < 2; ++i)
      #pragma unroll
      for (int j = 0; j < 4; ++j)
        acc[i][j] = __builtin_amdgcn_wmma_f32_16x16x32_bf16(
            false, a[i].v, false, b[j].v, (short)0, acc[i][j], false, false);
    tdm_wait();          // DMA for the next buffer must land before the barrier
    __syncthreads();
  }

  // Epilogue. C/D layout: VGPR r -> M = r + 8*(lane>>4); N = lane&15.
  const int rowb = m0 + wr * 32 + (lane >> 4) * 8;
  const int colb = n0 + wc * 64 + lm;
  #pragma unroll
  for (int i = 0; i < 2; ++i) {
    #pragma unroll
    for (int j = 0; j < 4; ++j) {
      const int col = colb + j * 16;
      const float bv = bias ? biasScale * bias[col] : 0.f;
      #pragma unroll
      for (int r = 0; r < 8; ++r) {
        const int row = rowb + i * 16 + r;
        const size_t idx = (size_t)row * ldc + col;
        float v = acc[i][j][r] + bv;
        if constexpr (ACCUM) v += Cf[idx];
        if constexpr (ACT == 1) v = v > 0.f ? v : 0.f;
        if constexpr (ACT == 2) v = 1.f / (1.f + __expf(-v));
        if constexpr (CBF) Cb[idx] = f2bf(v); else Cf[idx] = v;
      }
    }
  }
}

template <bool BT, int ACT, bool ACCUM, bool CBF>
static void gemm(hipStream_t st, const float* A, const bf16_t* B, void* C,
                 const float* bias, float bscale, int M, int N, int K,
                 int lda, long long sA, int ldb, long long sB1, long long sB2,
                 int bdiv, int ldc, long long sC, int batch) {
  dim3 g(N / BN, M / BM, batch);
  gemm_wmma_k<BT, ACT, ACCUM, CBF><<<g, dim3(256), 0, st>>>(
      A, B, C, bias, bscale, M, N, K, lda, sA, ldb, sB1, sB2, bdiv, ldc, sC);
}

// ---------------- small kernels ----------------
// Tiled convert+transpose: in fp32 [K,N] (per matrix) -> out bf16 [N,K].
// grid: (N/64, K/64, nMatrices); matrix strides K*N elements on both sides.
__global__ __launch_bounds__(256)
void cvt_tr_k(const float* __restrict__ in, bf16_t* __restrict__ out, int K, int N) {
  __shared__ bf16_t tile[64][72];
  const long long ms = (long long)K * N;
  in  += (long long)blockIdx.z * ms;
  out += (long long)blockIdx.z * ms;
  const int k0 = blockIdx.y * 64, n0 = blockIdx.x * 64;
  const int r = threadIdx.x >> 2;         // 0..63
  const int c = (threadIdx.x & 3) * 16;   // 0,16,32,48
  #pragma unroll
  for (int j = 0; j < 16; j += 4) {
    const float4 f = *(const float4*)(in + (size_t)(k0 + r) * N + n0 + c + j);
    tile[r][c + j + 0] = f2bf(f.x); tile[r][c + j + 1] = f2bf(f.y);
    tile[r][c + j + 2] = f2bf(f.z); tile[r][c + j + 3] = f2bf(f.w);
  }
  __syncthreads();
  union { bf16_t e[16]; uint4 q[2]; } vb;
  #pragma unroll
  for (int j = 0; j < 16; ++j) vb.e[j] = tile[c + j][r];
  bf16_t* op = out + (size_t)(n0 + r) * K + k0 + c;
  *(uint4*)(op)     = vb.q[0];
  *(uint4*)(op + 8) = vb.q[1];
}

__global__ void copy4_k(const float4* __restrict__ s, float4* __restrict__ d, int n4) {
  const int i = blockIdx.x * 256 + threadIdx.x;
  if (i < n4) d[i] = s[i];
}

__global__ __launch_bounds__(256)
void layernorm_k(const float* __restrict__ x, const float* __restrict__ g,
                 const float* __restrict__ b, float* __restrict__ y) {
  __shared__ float sm[256];
  const size_t row = blockIdx.x;
  const float* xr = x + row * Dm;
  float v[4];
  float s = 0.f;
  #pragma unroll
  for (int k = 0; k < 4; ++k) { v[k] = xr[threadIdx.x + k * 256]; s += v[k]; }
  s = block_red(s, sm, false);
  const float mean = s * (1.f / 1024.f);
  float s2 = 0.f;
  #pragma unroll
  for (int k = 0; k < 4; ++k) { const float d = v[k] - mean; s2 += d * d; }
  s2 = block_red(s2, sm, false);
  const float inv = rsqrtf(s2 * (1.f / 1024.f) + 1e-5f);
  float* yr = y + row * Dm;
  #pragma unroll
  for (int k = 0; k < 4; ++k) {
    const int c = threadIdx.x + k * 256;
    yr[c] = (v[k] - mean) * inv * g[c] + b[c];
  }
}

__global__ void posemb_k(float* __restrict__ pe) {
  const int i = blockIdx.x * 256 + threadIdx.x;   // FULL*Dm elems
  const int t = i >> 10, j = i & 1023;
  const int jj = j & 511;
  const float inv = __expf(-(float)(2 * jj) * (9.210340371976184f / 1024.f)); // 10000^(-2j/D)
  const float a = (float)(FULL - 1 - t) * inv;
  pe[i] = (j < 512) ? __sinf(a) : __cosf(a);
}

__global__ void adduv_k(const float* __restrict__ q, const float* __restrict__ u,
                        const float* __restrict__ v, float* __restrict__ qu,
                        float* __restrict__ qv) {
  const int i = blockIdx.x * 256 + threadIdx.x;
  const float qq = q[i];
  qu[i] = qq + u[i & 1023];
  qv[i] = qq + v[i & 1023];
}

// Fused rel_shift + mask + scale + softmax.  One block per (b,h,s) row of 1024.
// rel_shift: out[s,t] = padded[s*full + t + cur]; padded col 0 of each (full+1) row is 0.
__global__ __launch_bounds__(256)
void attn_softmax_k(float* __restrict__ att, const float* __restrict__ pos) {
  __shared__ float sm[256];
  const int rid = blockIdx.x;          // bh*CUR + s
  const int bh = rid >> 9;
  const int s  = rid & 511;
  float* row = att + (size_t)rid * FULL;
  const float* pb = pos + (size_t)bh * CUR * FULL;
  const float scale = 0.0883883476483184f;   // 1/sqrt(128)
  float v[4];
  float mx = -3.4e38f;
  #pragma unroll
  for (int k = 0; k < 4; ++k) {
    const int t = threadIdx.x + k * 256;
    const int g = s * FULL + t + CUR;
    const int ss = g / (FULL + 1);
    const int tt = g - ss * (FULL + 1);
    const float pv = (tt == 0) ? 0.f : pb[(size_t)ss * FULL + (tt - 1)];
    float lg = (row[t] + pv) * scale;
    if (t > s + MEM) lg = -1e30f;
    v[k] = lg;
    mx = fmaxf(mx, lg);
  }
  mx = block_red(mx, sm, true);
  float sum = 0.f;
  #pragma unroll
  for (int k = 0; k < 4; ++k) { v[k] = __expf(v[k] - mx); sum += v[k]; }
  sum = block_red(sum, sm, false);
  const float inv = 1.f / sum;
  #pragma unroll
  for (int k = 0; k < 4; ++k) row[threadIdx.x + k * 256] = v[k] * inv;
}

// out = (1-z)*x + z*tanh(tg + r*tu);  r,z already sigmoid'ed by GEMM epilogues.
__global__ void gru_k(const float* __restrict__ x, const float* __restrict__ tr,
                      const float* __restrict__ tz, const float* __restrict__ tg,
                      const float* __restrict__ tu, float* __restrict__ out) {
  const int i = blockIdx.x * 256 + threadIdx.x;
  const float r = tr[i], z = tz[i];
  const float h = tanhf(tg[i] + r * tu[i]);
  out[i] = (1.f - z) * x[i] + z * h;
}

// ======================================================================
extern "C" void kernel_launch(void* const* d_in, const int* in_sizes, int n_in,
                              void* d_out, int out_size, void* d_ws, size_t ws_size,
                              hipStream_t stream) {
  (void)n_in; (void)out_size; (void)ws_size;
  // ---- inputs (setup_inputs dict order) ----
  const float* x_in  = (const float*)d_in[0];
  const float* memin = (const float*)d_in[1];
  const float* bemb  = (const float*)d_in[3];
  const float* u_in  = (const float*)d_in[4];
  const float* v_in  = (const float*)d_in[5];
  const float* ln1g  = (const float*)d_in[6];
  const float* ln1b  = (const float*)d_in[7];
  const float* ln2g  = (const float*)d_in[8];
  const float* ln2b  = (const float*)d_in[9];
  const float* bkv   = (const float*)d_in[11];
  const float* bq    = (const float*)d_in[13];
  const float* bpos  = (const float*)d_in[15];
  const float* bproj = (const float*)d_in[17];
  const float* mb1   = (const float*)d_in[19];
  const float* mb2   = (const float*)d_in[21];
  const float* g1bg  = (const float*)d_in[28];
  const float* g2bg  = (const float*)d_in[35];

  // ---- workspace arena (~900 MB; assumes ws is sized generously) ----
  char* base = (char*)d_ws;
  size_t off = 0;
  auto mkalloc = [&](size_t bytes) -> char* {
    char* p = base + off;
    off = (off + bytes + 255) & ~(size_t)255;
    return p;
  };

  // bf16 + transposed ([N,K]) weight copies; re-done every call (graph-safe).
  static const int widx[19] = {2, 10, 12, 14, 16, 18, 20,
                               22, 23, 24, 25, 26, 27,   // g1 Wr Ur Wz Uz Wg Ug
                               29, 30, 31, 32, 33, 34};  // g2 ...
  bf16_t* wbf[36] = {};
  for (int w = 0; w < 19; ++w) {
    const int k = widx[w];
    const int N = (k == 10) ? 2048 : 1024;   // W_kv is [1024, 2048]
    const int nmat = in_sizes[k] / (1024 * N);
    wbf[k] = (bf16_t*)mkalloc((size_t)in_sizes[k] * sizeof(bf16_t));
    dim3 g(N / 64, 1024 / 64, nmat);
    cvt_tr_k<<<g, 256, 0, stream>>>((const float*)d_in[k], wbf[k], 1024, N);
  }

  const size_t SZC = (size_t)NCUR * Dm * 4;    // 32 MB
  float*  full = (float*)mkalloc((size_t)NFULL * Dm * 4);
  float*  x1   = (float*)mkalloc((size_t)NFULL * Dm * 4);
  bf16_t* kvb  = (bf16_t*)mkalloc((size_t)NFULL * 2 * Dm * 2);   // bf16 k|v
  float*  qb   = (float*)mkalloc(SZC);
  float*  qu   = (float*)mkalloc(SZC);
  float*  qv   = (float*)mkalloc(SZC);
  float*  pe   = (float*)mkalloc((size_t)FULL * Dm * 4);
  bf16_t* rb   = (bf16_t*)mkalloc((size_t)FULL * Dm * 2);        // bf16 r_
  float*  scon = (float*)mkalloc((size_t)BSz * Hh * CUR * FULL * 4);  // 256 MB (attn in-place)
  // spos (256 MB) is time-aliased with GRU/MLP temporaries (dead after softmax):
  char* regR  = mkalloc((size_t)BSz * Hh * CUR * FULL * 4);
  float* spos = (float*)regR;
  float* t1 = (float*)(regR + 0 * SZC);
  float* t2 = (float*)(regR + 1 * SZC);
  float* t3 = (float*)(regR + 2 * SZC);
  float* t4 = (float*)(regR + 3 * SZC);
  float* o1 = (float*)(regR + 4 * SZC);
  float* x2 = (float*)(regR + 5 * SZC);
  float* m1 = (float*)(regR + 6 * SZC);
  float* m2 = (float*)(regR + 7 * SZC);
  float* av = (float*)mkalloc(SZC);
  float* a1 = (float*)mkalloc(SZC);
  float* outb = (float*)d_out;   // running residual state lives directly in d_out

  const int NE  = NCUR * Dm;         // 8388608
  const int EG  = NE / 256;          // elementwise grid
  const size_t U = (size_t)Dm * Dm;

  // pos_emb (layer invariant)
  posemb_k<<<(FULL * Dm) / 256, 256, 0, stream>>>(pe);

  // embedding: out = relu(x @ W_emb + b_emb)
  gemm<true, 1, false, false>(stream, x_in, wbf[2], outb, bemb, 1.f,
                              NCUR, Dm, Dm, Dm, 0, Dm, 0, 0, 1, Dm, 0, 1);

  auto gru = [&](const float* xg, const float* yg, const bf16_t* Wr, const bf16_t* Ur,
                 const bf16_t* Wz, const bf16_t* Uz, const bf16_t* Wg, const bf16_t* Ug,
                 const float* bg, float* og) {
    gemm<true, 0, false, false>(stream, yg, Wr, t1, nullptr, 0.f, NCUR, Dm, Dm, Dm, 0, Dm, 0, 0, 1, Dm, 0, 1);
    gemm<true, 2, true , false>(stream, xg, Ur, t1, nullptr, 0.f, NCUR, Dm, Dm, Dm, 0, Dm, 0, 0, 1, Dm, 0, 1);
    gemm<true, 0, false, false>(stream, yg, Wz, t2, nullptr, 0.f, NCUR, Dm, Dm, Dm, 0, Dm, 0, 0, 1, Dm, 0, 1);
    gemm<true, 2, true , false>(stream, xg, Uz, t2, bg,   -1.f, NCUR, Dm, Dm, Dm, 0, Dm, 0, 0, 1, Dm, 0, 1);
    gemm<true, 0, false, false>(stream, yg, Wg, t3, nullptr, 0.f, NCUR, Dm, Dm, Dm, 0, Dm, 0, 0, 1, Dm, 0, 1);
    gemm<true, 0, false, false>(stream, xg, Ug, t4, nullptr, 0.f, NCUR, Dm, Dm, Dm, 0, Dm, 0, 0, 1, Dm, 0, 1);
    gru_k<<<EG, 256, 0, stream>>>(xg, t1, t2, t3, t4, og);
  };

  for (int i = 0; i < Lq; ++i) {
    const size_t iU = (size_t)i * U;
    // concat(memory[i], out) -> full
    copy4_k<<<EG / 4, 256, 0, stream>>>((const float4*)(memin + (size_t)i * NE), (float4*)full, NE / 4);
    copy4_k<<<EG / 4, 256, 0, stream>>>((const float4*)outb, (float4*)(full + NE), NE / 4);
    // x1 = LN1(full)
    layernorm_k<<<NFULL, 256, 0, stream>>>(full, ln1g + i * Dm, ln1b + i * Dm, x1);
    // kv projection -> bf16 (consumed only as WMMA B operands)
    gemm<true, 0, false, true>(stream, x1, wbf[10] + (size_t)i * Dm * 2048, kvb, bkv + i * 2048, 1.f,
                               NFULL, 2048, Dm, Dm, 0, Dm, 0, 0, 1, 2048, 0, 1);
    gemm<true, 0, false, false>(stream, x1 + (size_t)NE, wbf[12] + iU, qb, bq + i * Dm, 1.f,
                                NCUR, Dm, Dm, Dm, 0, Dm, 0, 0, 1, Dm, 0, 1);
    // r_ projection -> bf16
    gemm<true, 0, false, true>(stream, pe, wbf[14] + iU, rb, bpos + i * Dm, 1.f,
                               FULL, Dm, Dm, Dm, 0, Dm, 0, 0, 1, Dm, 0, 1);
    adduv_k<<<EG, 256, 0, stream>>>(qb, u_in, v_in, qu, qv);
    // content[z=b*H+h] = (q+u) @ k^T ; position = (q+v) @ r^T   (B = bf16 [N,K])
    gemm<true, 0, false, false>(stream, qu, kvb, scon, nullptr, 0.f,
                                CUR, FULL, HDd, BSz * HDm, HDd,
                                BSz * 2048, 2048, HDd, Hh, FULL, (long long)CUR * FULL, BSz * Hh);
    gemm<true, 0, false, false>(stream, qv, rb, spos, nullptr, 0.f,
                                CUR, FULL, HDd, BSz * HDm, HDd,
                                HDm, 0, HDd, Hh, FULL, (long long)CUR * FULL, BSz * Hh);
    // fused rel_shift + mask + softmax (scon becomes attn, in place)
    attn_softmax_k<<<BSz * Hh * CUR, 256, 0, stream>>>(scon, spos);
    // a_vec = attn @ v   (B = bf16 [K,N])
    gemm<false, 0, false, false>(stream, scon, kvb + HDm, av, nullptr, 0.f,
                                 CUR, HDd, FULL, FULL, (long long)CUR * FULL,
                                 BSz * 2048, 2048, HDd, Hh, BSz * HDm, HDd, BSz * Hh);
    // a1 = relu(a_vec @ W_proj + b_proj)
    gemm<true, 1, false, false>(stream, av, wbf[16] + iU, a1, bproj + i * Dm, 1.f,
                                NCUR, Dm, Dm, Dm, 0, Dm, 0, 0, 1, Dm, 0, 1);
    // GRU gate 1: o1 = gate(out, a1)
    gru(outb, a1, wbf[22] + iU, wbf[23] + iU, wbf[24] + iU, wbf[25] + iU,
        wbf[26] + iU, wbf[27] + iU, g1bg + i * Dm, o1);
    // MLP
    layernorm_k<<<NCUR, 256, 0, stream>>>(o1, ln2g + i * Dm, ln2b + i * Dm, x2);
    gemm<true, 1, false, false>(stream, x2, wbf[18] + iU, m1, mb1 + i * Dm, 1.f,
                                NCUR, Dm, Dm, Dm, 0, Dm, 0, 0, 1, Dm, 0, 1);
    gemm<true, 1, false, false>(stream, m1, wbf[20] + iU, m2, mb2 + i * Dm, 1.f,
                                NCUR, Dm, Dm, Dm, 0, Dm, 0, 0, 1, Dm, 0, 1);
    // GRU gate 2: out = gate(o1, m2)  (written straight into d_out)
    gru(o1, m2, wbf[29] + iU, wbf[30] + iU, wbf[31] + iU, wbf[32] + iU,
        wbf[33] + iU, wbf[34] + iU, g2bg + i * Dm, outb);
  }
}
